// TransposedLinearComponent_60559038874132
// MI455X (gfx1250) — matrix operand order repaired and checked by reading, not compiled, not run on hardware
//
#include <hip/hip_runtime.h>

// ---------------------------------------------------------------------------
// TransposedLinearComponent for MI455X (gfx1250), bf16 WMMA path.
//   b=1024, i=8, C=32, d=k=512, m=32
//   out[b,i,k] = sum_C mask[b,i,C] * sum_m A[i,C,k,m] * sum_d B[i,C,m,d]*x[b,i,d]
// Compute-bound (17.2 GFLOP vs ~3us HBM floor) -> v_wmma_f32_16x16x32_bf16
// (8x K per instruction vs f32 16x16x4 WMMA), f32 accumulation throughout.
// 256-thread blocks (8 waves) -> ~4 waves/SIMD for L2-latency hiding.
// ---------------------------------------------------------------------------

typedef __bf16 bf16_t;
typedef __attribute__((ext_vector_type(16))) bf16_t v16bf;
typedef __attribute__((ext_vector_type(8)))  bf16_t bf16x8;
typedef __attribute__((ext_vector_type(4)))  bf16_t bf16x4;
typedef __attribute__((ext_vector_type(8)))  float  v8f;
typedef __attribute__((ext_vector_type(4)))  float  f32x4;

#define B_TOT 1024
#define I_TOT 8
#define C_TOT 32
#define D_TOT 512
#define M_TOT 32

static constexpr int XROW = D_TOT + 8;   // padded LDS row stride (bf16 elems)
static constexpr int IROW = M_TOT + 8;   // padded inner row stride (bf16 elems)

// Assemble a 16-element bf16 operand from two 8-element (16B) chunks.
// gap = 8  -> 16 contiguous elems   (B-operand: one column, contiguous K)
// gap = 16 -> chunks at +0 and +16  (unused: A operands also load with gap = 8)
__device__ __forceinline__ v16bf load_op16(const bf16_t* p, int gap) {
    union { v16bf v; bf16x8 h[2]; } u;
    u.h[0] = *(const bf16x8*)(p);
    u.h[1] = *(const bf16x8*)(p + gap);
    return u.v;
}

__device__ __forceinline__ v8f wmma_bf16(v16bf a, v16bf b, v8f c) {
    return __builtin_amdgcn_wmma_f32_16x16x32_bf16(
        /*neg_a=*/false, a, /*neg_b=*/false, b,
        /*c_mod=*/(short)0, c, /*reuse_a=*/false, /*reuse_b=*/false);
}

// --------------------------- f32 -> bf16 pre-pass ---------------------------
__global__ void __launch_bounds__(256) cvt_f32_to_bf16(
        const float* __restrict__ in, bf16_t* __restrict__ out, int n4) {
    int idx = (blockIdx.x * 256 + threadIdx.x);
    if (idx < n4) {
        f32x4 v = *(const f32x4*)(in + idx * 4);
        bf16x4 o;
        o.x = (bf16_t)v.x; o.y = (bf16_t)v.y;
        o.z = (bf16_t)v.z; o.w = (bf16_t)v.w;
        *(bf16x4*)(out + idx * 4) = o;
    }
}

// ------------------------------- main kernel -------------------------------
// grid = (64 b-tiles, 8 i), block = 256 threads = 8 waves.
// Round r (4 rounds): wave w computes inner for C = 8r+w (GEMM1, K=512 in
// registers), masks+rounds to bf16 into LDS; then every wave runs its
// 64-column k-slice of GEMM2 over the 8 fresh C inners.
__global__ void __launch_bounds__(256) tlc_wmma_kernel(
        const bf16_t* __restrict__ xh,    // [b][i][d]       bf16
        const bf16_t* __restrict__ Ah,    // [i][C][k][m]    bf16
        const bf16_t* __restrict__ Bh,    // [i][C][m][d]    bf16
        const int*    __restrict__ mask,  // [b][i][C]       int32 (0/1)
        float*        __restrict__ out)   // [b][i][k]       f32
{
    __shared__ __align__(16) bf16_t xT[16 * XROW];        // x tile (16x512)
    __shared__ __align__(16) float  mT[16 * C_TOT];       // mask tile (float)
    __shared__ __align__(16) bf16_t innT[8][16 * IROW];   // masked inner, 8 C slots

    const int bt   = blockIdx.x;
    const int i    = blockIdx.y;
    const int b0   = bt * 16;
    const int tid  = threadIdx.x;
    const int lane = tid & 31;
    const int wave = tid >> 5;           // 0..7
    const int l16  = lane & 15;
    const int h    = lane >> 4;          // half-wave select (0/1)

    // ---- stage x tile in LDS (row-padded => conflict-free b128 reads) ----
    {
        const int row = tid >> 4;              // 0..15
        const int c0  = (tid & 15) * 32;       // 0..480
        const bf16_t* src = xh + ((size_t)(b0 + row) * I_TOT + i) * D_TOT + c0;
        bf16_t* dst = xT + row * XROW + c0;
        #pragma unroll
        for (int j = 0; j < 32; j += 8)
            *(bf16x8*)(dst + j) = *(const bf16x8*)(src + j);
    }
    // ---- stage mask tile as floats ----
    for (int j = tid; j < 16 * C_TOT; j += 256) {
        const int r = j >> 5, c = j & 31;
        mT[j] = (float)mask[((size_t)(b0 + r) * I_TOT + i) * C_TOT + c];
    }
    __syncthreads();

    v8f oacc[4];
    #pragma unroll
    for (int t = 0; t < 4; ++t) oacc[t] = (v8f)(0.0f);

    for (int r = 0; r < 4; ++r) {
        const int Cw = r * 8 + wave;

        // ===== GEMM1: inner[16x32] = Xtile[16x512] * B[i,Cw][32x512]^T =====
        v8f acc0 = (v8f)(0.0f), acc1 = (v8f)(0.0f);
        const bf16_t* Bbase = Bh + (size_t)(i * C_TOT + Cw) * M_TOT * D_TOT;
        #pragma unroll 4
        for (int dstp = 0; dstp < 16; ++dstp) {
            const int d0 = dstp * 32;
            // A operand: row l16, 16 contiguous K at h*16 from LDS
            v16bf aop = load_op16(xT + l16 * XROW + d0 + h * 16, 8);
            // B operands: column m = (tile*16 + l16), 16 contiguous d at h*16
            v16bf b0op = load_op16(Bbase + (size_t)(l16)      * D_TOT + d0 + h * 16, 8);
            v16bf b1op = load_op16(Bbase + (size_t)(16 + l16) * D_TOT + d0 + h * 16, 8);
            acc0 = wmma_bf16(aop, b0op, acc0);
            acc1 = wmma_bf16(aop, b1op, acc1);
        }

        // ===== mask + round to bf16 + stash inner in LDS =====
        {
            const int mrow0 = h * 8;             // C/D layout: rows v + 8*h
            bf16_t* ip = innT[wave];
            #pragma unroll
            for (int v = 0; v < 8; ++v) {
                const float mv = mT[(mrow0 + v) * C_TOT + Cw];
                ip[(mrow0 + v) * IROW + l16]      = (bf16_t)(acc0[v] * mv);
                ip[(mrow0 + v) * IROW + 16 + l16] = (bf16_t)(acc1[v] * mv);
            }
        }
        __syncthreads();

        // ===== GEMM2: out[:, kslice] += inner[16x32] * A[i,C][512x32]^T =====
        const int ksl = wave * 64;
        #pragma unroll
        for (int c = 0; c < 8; ++c) {
            const int Cc = r * 8 + c;
            v16bf aop2 = load_op16(innT[c] + l16 * IROW + h * 16, 8);
            const bf16_t* Abase = Ah + (size_t)(i * C_TOT + Cc) * D_TOT * M_TOT;
            #pragma unroll
            for (int t = 0; t < 4; ++t) {
                const int kcol = ksl + t * 16 + l16;
                v16bf bop = load_op16(Abase + (size_t)kcol * M_TOT + h * 16, 8);
                oacc[t] = wmma_bf16(aop2, bop, oacc[t]);
            }
        }
        __syncthreads();   // protect innT before next round overwrites it
    }

    // ===== epilogue: f32 stores, 64B contiguous per half-wave =====
    #pragma unroll
    for (int t = 0; t < 4; ++t) {
        const int kcol = wave * 64 + t * 16 + l16;
        #pragma unroll
        for (int v = 0; v < 8; ++v) {
            const int row = b0 + h * 8 + v;
            out[((size_t)row * I_TOT + i) * D_TOT + kcol] = oacc[t][v];
        }
    }
}

// ------------------------------- launcher ----------------------------------
extern "C" void kernel_launch(void* const* d_in, const int* in_sizes, int n_in,
                              void* d_out, int out_size, void* d_ws, size_t ws_size,
                              hipStream_t stream) {
    const float* x    = (const float*)d_in[0];   // [1024,8,512]
    const int*   mask = (const int*)  d_in[1];   // [1024,8,32]
    const float* A    = (const float*)d_in[2];   // [8,32,512,32]
    const float* B    = (const float*)d_in[3];   // [8,32,32,512]
    float*       out  = (float*)d_out;

    constexpr int XN = B_TOT * I_TOT * D_TOT;        // 4,194,304
    constexpr int WN = I_TOT * C_TOT * D_TOT * M_TOT;// 4,194,304

    bf16_t* xh = (bf16_t*)d_ws;
    bf16_t* Ah = xh + XN;
    bf16_t* Bh = Ah + WN;

    cvt_f32_to_bf16<<<XN / 4 / 256, 256, 0, stream>>>(x, xh, XN / 4);
    cvt_f32_to_bf16<<<WN / 4 / 256, 256, 0, stream>>>(A, Ah, WN / 4);
    cvt_f32_to_bf16<<<WN / 4 / 256, 256, 0, stream>>>(B, Bh, WN / 4);

    tlc_wmma_kernel<<<dim3(B_TOT / 16, I_TOT), 256, 0, stream>>>(xh, Ah, Bh, mask, out);
}
